// BiLSTMCRF_54932631715888
// MI455X (gfx1250) — matrix operand order repaired and checked by reading, not compiled, not run on hardware
//
#include <hip/hip_runtime.h>

// ---------------------------------------------------------------------------
// BiLSTM-CRF loss for MI455X (gfx1250, wave32, WMMA bf16 16x16x32)
// ---------------------------------------------------------------------------
#define B_     128
#define T_     512
#define E_     256
#define H_     256
#define K_     20
#define GATES_ 1024   // 4H
#define KCOMB_ 512    // H + E (combined recurrent+input GEMM K dim)
#define HID2_  512    // 2H
#define NPAD_  32     // emit columns padded 20 -> 32

typedef __attribute__((ext_vector_type(16))) __bf16 v16bf;
typedef __attribute__((ext_vector_type(8)))  float  v8f;

union AFrag { v16bf v; unsigned u[8]; };

#if defined(__has_builtin)
# if __has_builtin(__builtin_amdgcn_global_load_async_to_lds_b128)
#  define ASYNC_X_STAGE 1
# endif
# if __has_builtin(__builtin_amdgcn_s_wait_asynccnt)
#  define HAVE_WAIT_ASYNC 1
# endif
#endif

#if ASYNC_X_STAGE
// Exact parameter types per hipcc diagnostic: v4i32* in global (AS1) /
// LDS (AS3) address spaces. Integer casts: generic global VA == AS1 VA;
// generic LDS VA carries the 32-bit LDS offset in its low DWORD (ISA 10.2).
typedef __attribute__((__vector_size__(16))) int v4i_;
typedef __attribute__((address_space(1))) v4i_ gv4i_;
typedef __attribute__((address_space(3))) v4i_ lv4i_;
#endif

static __device__ __forceinline__ unsigned short f2bf(float f) {
  unsigned u = __float_as_uint(f);
  return (unsigned short)((u + 0x7FFFu + ((u >> 16) & 1u)) >> 16);
}
// Fast nonlinearities: hardware v_rcp_f32 instead of IEEE division chains.
static __device__ __forceinline__ float sigm(float x) {
  return __builtin_amdgcn_rcpf(1.f + __expf(-x));
}
static __device__ __forceinline__ float tanhf_(float x) {
  return 1.f - 2.f * __builtin_amdgcn_rcpf(__expf(2.f * x) + 1.f);
}

// ---------------------------------------------------------------------------
// 1) Embedding gather + f32->bf16, layout xq[t][b][e]
// ---------------------------------------------------------------------------
__global__ void __launch_bounds__(256) embed_gather_kernel(
    const int* __restrict__ sentences, const float* __restrict__ embedding,
    unsigned short* __restrict__ xq) {
  int idx = blockIdx.x * 256 + threadIdx.x;     // one thread = 4 elements
  if (idx >= (T_ * B_ * E_) / 4) return;
  int e4 = idx & (E_ / 4 - 1);                  // 0..63
  int tb = idx >> 6;                            // t*B + b
  int t = tb >> 7, b = tb & (B_ - 1);
  int tok = sentences[b * T_ + t];
  float4 v = *(const float4*)(embedding + (size_t)tok * E_ + e4 * 4);
  unsigned short* dst = xq + (size_t)tb * E_ + e4 * 4;
  dst[0] = f2bf(v.x); dst[1] = f2bf(v.y); dst[2] = f2bf(v.z); dst[3] = f2bf(v.w);
}

// ---------------------------------------------------------------------------
// 2) Swizzle weights into per-lane WMMA B-fragment layout.
//    Fragment = 32(K) x 16(N); lane L holds N=L&15, K = (L>>4)*16 + i (i<16),
//    i.e. 16 contiguous bf16 (one aligned 32B block) per lane per fragment.
//    Combined W: k<H -> w_hh[n][k]; k>=H -> w_ih[n][k-H].
// ---------------------------------------------------------------------------
__global__ void __launch_bounds__(256) swizzle_w_kernel(
    const float* __restrict__ w_ih_f, const float* __restrict__ w_hh_f,
    const float* __restrict__ w_ih_b, const float* __restrict__ w_hh_b,
    const float* __restrict__ w_out,
    unsigned short* __restrict__ Wf, unsigned short* __restrict__ Wb,
    unsigned short* __restrict__ Wo) {
  int idx = blockIdx.x * 256 + threadIdx.x;
  const int WSZ = KCOMB_ * GATES_;              // 524288 elems per direction
  if (idx < 2 * WSZ) {
    int dir = (idx >= WSZ) ? 1 : 0;
    int rem = idx - dir * WSZ;
    int frag = rem >> 9, within = rem & 511;
    int lane = within >> 4, i = within & 15;
    int ntile = frag >> 4, kf = frag & 15;      // frag = ntile*16 + kf
    int k = kf * 32 + ((lane >> 4) << 4) + i;
    int n = ntile * 16 + (lane & 15);
    const float* wih = dir ? w_ih_b : w_ih_f;
    const float* whh = dir ? w_hh_b : w_hh_f;
    float val = (k < H_) ? whh[n * H_ + k] : wih[n * E_ + (k - H_)];
    (dir ? Wb : Wf)[rem] = f2bf(val);
  } else {
    int rem = idx - 2 * WSZ;
    if (rem >= KCOMB_ * NPAD_) return;          // 16384
    int frag = rem >> 9, within = rem & 511;
    int lane = within >> 4, i = within & 15;
    int ntile = frag >> 4, kf = frag & 15;
    int k = kf * 32 + ((lane >> 4) << 4) + i;
    int n = ntile * 16 + (lane & 15);
    float val = (n < K_) ? w_out[n * HID2_ + k] : 0.f;
    Wo[rem] = f2bf(val);
  }
}

// ---------------------------------------------------------------------------
// 3) Persistent masked LSTM. Grid = 16 WGs (2 dirs x 8 batch-chunks of 16),
//    512 threads = 16 waves per WG. Wave w owns hidden tile ht=w and its 4
//    gate tiles {ht, 16+ht, 32+ht, 48+ht} -> 4 f32 accumulator tiles, so the
//    full i/f/g/o cell update is done in registers. W fragments are streamed
//    from L2 each step; an opaque 32-bit offset (laundered through asm) stops
//    LICM from hoisting the weight loads while keeping the base pointer's
//    global address space known (global_load_b128, not flat).
// ---------------------------------------------------------------------------
__global__ void __launch_bounds__(512) lstm_kernel(
    const unsigned short* __restrict__ xq,
    const unsigned short* __restrict__ Wf,
    const unsigned short* __restrict__ Wb,
    const float* __restrict__ b_f, const float* __restrict__ b_b,
    const int* __restrict__ sentences,
    unsigned short* __restrict__ hid) {
  const int dir = blockIdx.x >> 3;
  const int b0  = (blockIdx.x & 7) * 16;
  const unsigned short* __restrict__ Wd = dir ? Wb : Wf;
  const float* __restrict__ bias = dir ? b_b : b_f;

  __shared__ __align__(16) unsigned short a_buf[16][KCOMB_]; // [row][h(256)|x(256)]
  __shared__ float maskrow[16];

  const int tid = threadIdx.x;
  const int wid = tid >> 5, lane = tid & 31;
  const int ml = lane & 15, lh = lane >> 4;
  const int ht = wid;                    // hidden tile of this wave (0..15)

  { // h := 0 (16 rows x 256 cols, 8 bf16 per thread)
    int row = tid >> 5, col = (tid & 31) * 8;
    *(uint4*)&a_buf[row][col] = make_uint4(0, 0, 0, 0);
  }

  v8f creg = {};                         // c state tile (16x16 f32)
  float bi, bff, bg, bo;
  {
    int n0 = ht * 16 + ml;
    bi  = bias[0 * H_ + n0];
    bff = bias[1 * H_ + n0];
    bg  = bias[2 * H_ + n0];
    bo  = bias[3 * H_ + n0];
  }

  for (int step = 0; step < T_; ++step) {
    const int t = dir ? (T_ - 1 - step) : step;
    __syncthreads();                                   // prev h writes visible
    {                                                  // stage x tile (16x256 bf16)
      int r = tid >> 5, ch = tid & 31;
      const unsigned short* src = xq + ((size_t)t * B_ + (b0 + r)) * E_ + ch * 8;
      unsigned short* dst = &a_buf[r][H_ + ch * 8];
#if ASYNC_X_STAGE
      __builtin_amdgcn_global_load_async_to_lds_b128(
          (gv4i_*)(unsigned long long)src,
          (lv4i_*)(unsigned)(unsigned long long)dst, 0, 0);
# if HAVE_WAIT_ASYNC
      __builtin_amdgcn_s_wait_asynccnt(0);
# else
      asm volatile("s_wait_asynccnt 0" ::: "memory");
# endif
#else
      *(uint4*)dst = *(const uint4*)src;
#endif
    }
    if (tid < 16) maskrow[tid] = (sentences[(b0 + tid) * T_ + t] != 0) ? 1.f : 0.f;
    __syncthreads();

    // Opaque per-step offset: keeps weight loads inside the time loop (so
    // they stream from L2 instead of being hoisted + spilled to scratch)
    // while preserving the provably-global base pointer.
    unsigned woff = 0;
    asm volatile("" : "+v"(woff));
    const unsigned short* Wdt = Wd + woff;

    v8f acc[4] = {};                                   // i, f, g, o tiles
    for (int kf = 0; kf < 16; ++kf) {
      AFrag a;                                         // A frag (16x32 bf16) from LDS
      #pragma unroll
      for (int v = 0; v < 8; ++v) {
        int k = kf * 32 + (lh << 4) + (v << 1);
        a.u[v] = *(const unsigned*)&a_buf[ml][k];
      }
      __builtin_prefetch(Wdt + (((size_t)ht * 16 + kf + 1) << 9) + lane * 16, 0, 1);
      #pragma unroll
      for (int g = 0; g < 4; ++g) {
        int ntile = g * 16 + ht;
        const v16bf* bp =
            (const v16bf*)(Wdt + (((size_t)ntile * 16 + kf) << 9) + lane * 16);
        acc[g] = __builtin_amdgcn_wmma_f32_16x16x32_bf16(
            false, a.v, false, *bp, (short)0, acc[g], false, false);
      }
    }
    __syncthreads();                                   // all GEMM reads of a_buf done

    const int col = ht * 16 + ml;
    #pragma unroll
    for (int r = 0; r < 8; ++r) {
      int M = r + (lh << 3);
      float m  = maskrow[M];
      float iv = sigm  (acc[0][r] + bi);
      float fv = sigm  (acc[1][r] + bff);
      float gv = tanhf_(acc[2][r] + bg);
      float ov = sigm  (acc[3][r] + bo);
      float c0 = creg[r];
      float cn = fv * c0 + iv * gv;
      float hn = ov * tanhf_(cn);
      bool msk = m > 0.5f;
      creg[r] = msk ? cn : c0;
      if (msk) a_buf[M][col] = f2bf(hn);               // h frozen when masked
      hid[((size_t)t * B_ + (b0 + M)) * HID2_ + dir * H_ + col] =
          f2bf(msk ? hn : 0.f);                        // out zeroed when masked
    }
  }
}

// ---------------------------------------------------------------------------
// 4) emit = hid @ w_out^T + b_out  (M=65536, K=512, N=32 padded). WMMA bf16.
// ---------------------------------------------------------------------------
__global__ void __launch_bounds__(256) emit_kernel(
    const unsigned short* __restrict__ hid,
    const unsigned short* __restrict__ Wo,
    const float* __restrict__ b_out,
    float* __restrict__ emit) {
  const int tid = threadIdx.x, wid = tid >> 5, lane = tid & 31;
  const int ml = lane & 15, lh = lane >> 4;
  const size_t M0 = ((size_t)blockIdx.x * 8 + wid) * 16;
  const unsigned short* arow = hid + (M0 + ml) * HID2_;
  v8f acc0 = {}, acc1 = {};
  for (int kf = 0; kf < 16; ++kf) {
    AFrag a;
    #pragma unroll
    for (int v = 0; v < 8; ++v) {
      int k = kf * 32 + (lh << 4) + (v << 1);
      a.u[v] = *(const unsigned*)(arow + k);
    }
    const v16bf* bp0 = (const v16bf*)(Wo + ((size_t)(kf)      << 9) + lane * 16);
    const v16bf* bp1 = (const v16bf*)(Wo + ((size_t)(16 + kf) << 9) + lane * 16);
    acc0 = __builtin_amdgcn_wmma_f32_16x16x32_bf16(false, a.v, false, *bp0, (short)0, acc0, false, false);
    acc1 = __builtin_amdgcn_wmma_f32_16x16x32_bf16(false, a.v, false, *bp1, (short)0, acc1, false, false);
  }
  float bias0 = (ml < K_)      ? b_out[ml]      : 0.f;
  float bias1 = (16 + ml < K_) ? b_out[16 + ml] : 0.f;
  #pragma unroll
  for (int r = 0; r < 8; ++r) {
    size_t M = M0 + r + (lh << 3);
    emit[M * NPAD_ + ml]      = acc0[r] + bias0;
    emit[M * NPAD_ + 16 + ml] = acc1[r] + bias1;
  }
}

// ---------------------------------------------------------------------------
// 5) Gold-path score: sum of unary + transition over valid steps.
// ---------------------------------------------------------------------------
__global__ void __launch_bounds__(128) score_kernel(
    const int* __restrict__ sentences, const int* __restrict__ tags,
    const float* __restrict__ emit, const float* __restrict__ trans,
    float* __restrict__ ts_out) {
  int b = threadIdx.x;
  if (b >= B_) return;
  float ts = 0.f; int prev = 0;
  for (int t = 0; t < T_; ++t) {
    int tg = tags[b * T_ + t];
    float u = emit[((size_t)t * B_ + b) * NPAD_ + tg];
    float sc = u + (t > 0 ? trans[prev * K_ + tg] : 0.f);
    if (sentences[b * T_ + t] != 0) ts += sc;
    prev = tg;
  }
  ts_out[b] = ts;
}

// ---------------------------------------------------------------------------
// 6) CRF forward scan: one wave32 per batch row, lane = state (20 of 32),
//    online log-sum-exp with __shfl broadcasts of alpha.
// ---------------------------------------------------------------------------
__global__ void __launch_bounds__(256) crf_kernel(
    const int* __restrict__ sentences,
    const float* __restrict__ emit,
    const float* __restrict__ trans,
    const float* __restrict__ ts_in,
    float* __restrict__ out) {
  __shared__ float trS[K_][32];
  int tid = threadIdx.x;
  for (int idx = tid; idx < K_ * 32; idx += 256) {
    int i = idx >> 5, j = idx & 31;
    trS[i][j] = (j < K_) ? trans[i * K_ + j] : -1.0e30f;
  }
  __syncthreads();
  int wid = tid >> 5, lane = tid & 31;
  int b = blockIdx.x * 8 + wid;
  float d = (lane < K_) ? emit[(size_t)b * NPAD_ + lane] : -1.0e30f;  // t = 0
  for (int t = 1; t < T_; ++t) {
    float e = (lane < K_) ? emit[((size_t)t * B_ + b) * NPAD_ + lane] : 0.f;
    bool msk = sentences[b * T_ + t] != 0;
    float m = -3.0e38f, s = 0.f;
    #pragma unroll
    for (int i = 0; i < K_; ++i) {
      float di = __shfl(d, i, 32);
      float v = di + trS[i][lane];
      if (v > m) { s = s * __expf(m - v) + 1.f; m = v; }
      else       { s += __expf(v - m); }
    }
    float nd = m + __logf(s) + e;
    d = msk ? nd : d;
  }
  float m = -3.0e38f, s = 0.f;
  #pragma unroll
  for (int i = 0; i < K_; ++i) {
    float di = __shfl(d, i, 32);
    if (di > m) { s = s * __expf(m - di) + 1.f; m = di; }
    else        { s += __expf(di - m); }
  }
  float logz = m + __logf(s);
  if (lane == 0) out[b] = -(ts_in[b] - logz);
}

// ---------------------------------------------------------------------------
extern "C" void kernel_launch(void* const* d_in, const int* in_sizes, int n_in,
                              void* d_out, int out_size, void* d_ws, size_t ws_size,
                              hipStream_t stream) {
  (void)in_sizes; (void)n_in; (void)out_size; (void)ws_size;
  const int*   sentences = (const int*)  d_in[0];
  const int*   tags      = (const int*)  d_in[1];
  const float* embedding = (const float*)d_in[3];
  const float* w_ih_f    = (const float*)d_in[4];
  const float* w_hh_f    = (const float*)d_in[5];
  const float* b_f       = (const float*)d_in[6];
  const float* w_ih_b    = (const float*)d_in[7];
  const float* w_hh_b    = (const float*)d_in[8];
  const float* b_b       = (const float*)d_in[9];
  const float* w_out     = (const float*)d_in[10];
  const float* b_out     = (const float*)d_in[11];
  const float* trans     = (const float*)d_in[12];
  float* out = (float*)d_out;
  char*  ws  = (char*)d_ws;

  size_t off = 0;
  unsigned short* xq  = (unsigned short*)(ws + off); off += (size_t)T_ * B_ * E_ * 2;      // 32 MB
  unsigned short* Wf  = (unsigned short*)(ws + off); off += (size_t)KCOMB_ * GATES_ * 2;   // 1 MB
  unsigned short* Wb  = (unsigned short*)(ws + off); off += (size_t)KCOMB_ * GATES_ * 2;   // 1 MB
  unsigned short* Wo  = (unsigned short*)(ws + off); off += (size_t)KCOMB_ * NPAD_ * 2;    // 32 KB
  unsigned short* hid = (unsigned short*)(ws + off); off += (size_t)T_ * B_ * HID2_ * 2;   // 64 MB
  float*          emt = (float*)(ws + off);          off += (size_t)T_ * B_ * NPAD_ * 4;   // 8 MB
  float*          ts  = (float*)(ws + off);          off += (size_t)B_ * 4;

  embed_gather_kernel<<<(T_ * B_ * E_ / 4 + 255) / 256, 256, 0, stream>>>(sentences, embedding, xq);
  swizzle_w_kernel<<<(2 * KCOMB_ * GATES_ + KCOMB_ * NPAD_ + 255) / 256, 256, 0, stream>>>(
      w_ih_f, w_hh_f, w_ih_b, w_hh_b, w_out, Wf, Wb, Wo);
  lstm_kernel<<<16, 512, 0, stream>>>(xq, Wf, Wb, b_f, b_b, sentences, hid);
  emit_kernel<<<(T_ * B_ / 16) / 8, 256, 0, stream>>>(hid, Wo, b_out, emt);
  score_kernel<<<1, 128, 0, stream>>>(sentences, tags, emt, trans, ts);
  crf_kernel<<<B_ / 8, 256, 0, stream>>>(sentences, emt, trans, ts, out);
}
